// MoELayer_64183991271506
// MI455X (gfx1250) — matrix operand order repaired and checked by reading, not compiled, not run on hardware
//
#include <hip/hip_runtime.h>

// MoE layer for MI455X (gfx1250), wave32.
// WMMA bf16 (f32 accum) + async-to-LDS staging + double-buffered pipeline.
// Shapes: B=4 S=4096 D=1024 H=512 E=16 K=2, T=16384.

#define BB 4
#define SS 4096
#define DD 1024
#define HH 512
#define EE 16
#define TT (BB * SS)

typedef __attribute__((ext_vector_type(16))) __bf16 v16bf;
typedef __attribute__((ext_vector_type(8)))  float  v8f;

#define BM 128
#define BN 64
#define BK 32
#define LDT 40   // padded LDS stride (bf16 elements); row = 64B data + 16B pad

// ---------- scalar helpers ----------
__device__ __forceinline__ unsigned short f2bf(float f) {
  union { float f; unsigned u; } x; x.f = f;
  unsigned r = x.u + 0x7fffu + ((x.u >> 16) & 1u);   // round-to-nearest-even
  return (unsigned short)(r >> 16);
}
__device__ __forceinline__ float bf2f(unsigned short h) {
  union { unsigned u; float f; } x; x.u = ((unsigned)h) << 16;
  return x.f;
}
__device__ __forceinline__ float silu_f(float v) {
  return v * __builtin_amdgcn_rcpf(1.0f + __expf(-v));
}

// ---------- gfx1250 async copy: 16B global -> LDS per lane (ASYNCcnt) ----------
__device__ __forceinline__ void cp_async16(void* lds, const void* g) {
  unsigned ldsaddr = (unsigned)(unsigned long long)lds;   // flat low 32 bits = LDS addr
  asm volatile("global_load_async_to_lds_b128 %0, %1, off"
               :: "v"(ldsaddr), "v"(g) : "memory");
}
__device__ __forceinline__ void cp_async_wait() {
  asm volatile("s_wait_asynccnt 0x0" ::: "memory");
}

// ---------- WMMA fragment helpers (ISA 7.12.2 layouts, wave32) ----------
__device__ __forceinline__ v16bf ld_fragA(const unsigned short* As, int mBase, int lane) {
  int m = mBase + (lane & 15);
  int half = lane >> 4;
  union { v16bf v; unsigned u[8]; } f;
  const unsigned short* row = As + m * LDT;
#pragma unroll
  for (int i = 0; i < 4; ++i) {
    int k = half * 8 + i * 2;
    f.u[i] = *(const unsigned*)(row + k);
  }
#pragma unroll
  for (int i = 0; i < 4; ++i) {
    int k = 16 + half * 8 + i * 2;
    f.u[4 + i] = *(const unsigned*)(row + k);
  }
  return f.v;
}
__device__ __forceinline__ v16bf ld_fragB(const unsigned short* Bs, int nBase, int lane) {
  int n = nBase + (lane & 15);
  int half = lane >> 4;
  union { v16bf v; unsigned u[8]; } f;
  const unsigned short* col = Bs + n * LDT;
#pragma unroll
  for (int i = 0; i < 4; ++i) {
    int k = half * 8 + i * 2;
    f.u[i] = *(const unsigned*)(col + k);
  }
#pragma unroll
  for (int i = 0; i < 4; ++i) {
    int k = 16 + half * 8 + i * 2;
    f.u[4 + i] = *(const unsigned*)(col + k);
  }
  return f.v;
}
__device__ __forceinline__ v8f wmma_bf16(v16bf a, v16bf b, v8f c) {
  return __builtin_amdgcn_wmma_f32_16x16x32_bf16(false, a, false, b, (short)0, c, false, false);
}

// B tile from fp32 weight [K x N] row-major (ld = row stride); stored transposed [N][K].
__device__ __forceinline__ void load_B_f32(unsigned short* Bs, const float* w,
                                           int kb, int nBase, int ld, int tid) {
#pragma unroll
  for (int j = 0; j < 2; ++j) {
    int lin = tid + j * 256;
    int k   = lin >> 4;
    int n4  = (lin & 15) * 4;
    const float4 v = *(const float4*)(w + (size_t)(kb + k) * ld + nBase + n4);
    Bs[(n4 + 0) * LDT + k] = f2bf(v.x);
    Bs[(n4 + 1) * LDT + k] = f2bf(v.y);
    Bs[(n4 + 2) * LDT + k] = f2bf(v.z);
    Bs[(n4 + 3) * LDT + k] = f2bf(v.w);
  }
}

// Async A tile: 128 rows x 64B (BK=32 bf16). Wave w stages rows [w*16, w*16+16).
__device__ __forceinline__ void stage_A_async_dense(unsigned short* As,
                                                    const unsigned short* src,
                                                    size_t rowBase, int ldRow,
                                                    int kb, int wave, int lane) {
  int r0 = wave * 16 + (lane >> 2);
  int c  = (lane & 3) * 16;
  const char* s0 = (const char*)(src + (rowBase + (size_t)r0) * ldRow + kb) + c;
  const char* s1 = (const char*)(src + (rowBase + (size_t)(r0 + 8)) * ldRow + kb) + c;
  cp_async16((char*)As + r0 * (LDT * 2) + c, s0);
  cp_async16((char*)As + (r0 + 8) * (LDT * 2) + c, s1);
}
__device__ __forceinline__ void stage_A_async_gather(unsigned short* As,
                                                     const unsigned short* src,
                                                     const int* sEnt, int entShift,
                                                     int ldRow, int kb,
                                                     const char* zbuf,
                                                     int wave, int lane) {
  int r0 = wave * 16 + (lane >> 2);
  int c  = (lane & 3) * 16;
#pragma unroll
  for (int h = 0; h < 2; ++h) {
    int r = r0 + h * 8;
    int ent = sEnt[r];
    const char* s = (ent >= 0)
        ? (const char*)(src + (size_t)(ent >> entShift) * ldRow + kb) + c
        : zbuf + c;
    cp_async16((char*)As + r * (LDT * 2) + c, s);
  }
}

// ---------- kernel: x -> bf16 (one-shot) ----------
__global__ __launch_bounds__(256) void k_cvt_x(const float* __restrict__ x,
                                               unsigned short* __restrict__ xb) {
  size_t i = ((size_t)blockIdx.x * 256 + threadIdx.x) * 8;
  float4 a = *(const float4*)(x + i);
  float4 b = *(const float4*)(x + i + 4);
  uint4 o;
  o.x = (unsigned)f2bf(a.x) | ((unsigned)f2bf(a.y) << 16);
  o.y = (unsigned)f2bf(a.z) | ((unsigned)f2bf(a.w) << 16);
  o.z = (unsigned)f2bf(b.x) | ((unsigned)f2bf(b.y) << 16);
  o.w = (unsigned)f2bf(b.z) | ((unsigned)f2bf(b.w) << 16);
  *(uint4*)(xb + i) = o;
}

// ---------- kernel: router + top-2 ----------
__global__ __launch_bounds__(256) void k_router(const float* __restrict__ x,
                                                const float* __restrict__ wr,
                                                float* __restrict__ logits_out,
                                                int* __restrict__ cnt,
                                                int* __restrict__ ids,
                                                float* __restrict__ gwe) {
  int t = blockIdx.x * 256 + threadIdx.x;
  if (t >= TT) return;
  const float* xr = x + (size_t)t * DD;
  float acc[EE];
#pragma unroll
  for (int e = 0; e < EE; ++e) acc[e] = 0.0f;
  for (int d = 0; d < DD; d += 4) {
    float4 xv = *(const float4*)(xr + d);
    const float* w0 = wr + d * EE;
#pragma unroll
    for (int e = 0; e < EE; ++e) {
      acc[e] = fmaf(xv.x, w0[e], acc[e]);
      acc[e] = fmaf(xv.y, w0[EE + e], acc[e]);
      acc[e] = fmaf(xv.z, w0[2 * EE + e], acc[e]);
      acc[e] = fmaf(xv.w, w0[3 * EE + e], acc[e]);
    }
  }
  float* lo = logits_out + (size_t)t * EE;
#pragma unroll
  for (int e = 0; e < EE; ++e) lo[e] = acc[e];
  float m = acc[0];
#pragma unroll
  for (int e = 1; e < EE; ++e) m = fmaxf(m, acc[e]);
  float p[EE]; float s = 0.0f;
#pragma unroll
  for (int e = 0; e < EE; ++e) { p[e] = __expf(acc[e] - m); s += p[e]; }
  float inv = 1.0f / s;
  int i0 = 0; float w0v = -1.0f;
#pragma unroll
  for (int e = 0; e < EE; ++e) { float pe = p[e] * inv; p[e] = pe; if (pe > w0v) { w0v = pe; i0 = e; } }
  int i1 = 0; float w1v = -1.0f;
#pragma unroll
  for (int e = 0; e < EE; ++e) { if (e != i0 && p[e] > w1v) { w1v = p[e]; i1 = e; } }
  float gs = 1.0f / (w0v + w1v);
  int p0 = atomicAdd(&cnt[i0], 1);
  ids[i0 * TT + p0] = 2 * t;
  gwe[2 * t] = w0v * gs;
  int p1 = atomicAdd(&cnt[i1], 1);
  ids[i1 * TT + p1] = 2 * t + 1;
  gwe[2 * t + 1] = w1v * gs;
}

// ---------- kernel: shared expert stage 1 (double-buffered) ----------
__global__ __launch_bounds__(256) void k_shared_h(const unsigned short* __restrict__ xb,
                                                  const float* __restrict__ ws1,
                                                  const float* __restrict__ ws2,
                                                  unsigned short* __restrict__ hs) {
  __shared__ unsigned short As[2][BM * LDT];
  __shared__ unsigned short Bs1[2][BN * LDT];
  __shared__ unsigned short Bs2[2][BN * LDT];
  const int tid = threadIdx.x, lane = tid & 31, wave = tid >> 5;
  const int wm = (wave & 3) * 32, wn = (wave >> 2) * 32;
  const size_t rowBase = (size_t)blockIdx.x * BM;
  const int nBase = blockIdx.y * BN;
  v8f z = {0.f, 0.f, 0.f, 0.f, 0.f, 0.f, 0.f, 0.f};
  v8f a1[2][2], a2[2][2];
#pragma unroll
  for (int i = 0; i < 2; ++i)
#pragma unroll
    for (int j = 0; j < 2; ++j) { a1[i][j] = z; a2[i][j] = z; }

  stage_A_async_dense(As[0], xb, rowBase, DD, 0, wave, lane);
  load_B_f32(Bs1[0], ws1, 0, nBase, HH, tid);
  load_B_f32(Bs2[0], ws2, 0, nBase, HH, tid);
  const int nsteps = DD / BK;
  for (int s = 0; s < nsteps; ++s) {
    const int cur = s & 1;
    cp_async_wait();
    __syncthreads();
    if (s + 1 < nsteps) {
      const int kb = (s + 1) * BK;
      stage_A_async_dense(As[cur ^ 1], xb, rowBase, DD, kb, wave, lane);
      load_B_f32(Bs1[cur ^ 1], ws1, kb, nBase, HH, tid);
      load_B_f32(Bs2[cur ^ 1], ws2, kb, nBase, HH, tid);
    }
    v16bf fa0 = ld_fragA(As[cur], wm, lane);
    v16bf fa1 = ld_fragA(As[cur], wm + 16, lane);
    v16bf b10 = ld_fragB(Bs1[cur], wn, lane);
    v16bf b11 = ld_fragB(Bs1[cur], wn + 16, lane);
    v16bf b20 = ld_fragB(Bs2[cur], wn, lane);
    v16bf b21 = ld_fragB(Bs2[cur], wn + 16, lane);
    a1[0][0] = wmma_bf16(fa0, b10, a1[0][0]);
    a1[0][1] = wmma_bf16(fa0, b11, a1[0][1]);
    a1[1][0] = wmma_bf16(fa1, b10, a1[1][0]);
    a1[1][1] = wmma_bf16(fa1, b11, a1[1][1]);
    a2[0][0] = wmma_bf16(fa0, b20, a2[0][0]);
    a2[0][1] = wmma_bf16(fa0, b21, a2[0][1]);
    a2[1][0] = wmma_bf16(fa1, b20, a2[1][0]);
    a2[1][1] = wmma_bf16(fa1, b21, a2[1][1]);
  }
  const int n_l = lane & 15, mh = (lane >> 4) * 8;
#pragma unroll
  for (int im = 0; im < 2; ++im)
#pragma unroll
    for (int in = 0; in < 2; ++in)
#pragma unroll
      for (int i = 0; i < 8; ++i) {
        float h = silu_f(a1[im][in][i]) * a2[im][in][i];
        size_t gm = rowBase + wm + im * 16 + mh + i;
        int gn = nBase + wn + in * 16 + n_l;
        hs[gm * HH + gn] = f2bf(h);
      }
}

// ---------- kernel: shared expert stage 2 (double-buffered) ----------
__global__ __launch_bounds__(256) void k_shared_out(const unsigned short* __restrict__ hs,
                                                    const float* __restrict__ wso,
                                                    float* __restrict__ out) {
  __shared__ unsigned short As[2][BM * LDT];
  __shared__ unsigned short Bs[2][BN * LDT];
  const int tid = threadIdx.x, lane = tid & 31, wave = tid >> 5;
  const int wm = (wave & 3) * 32, wn = (wave >> 2) * 32;
  const size_t rowBase = (size_t)blockIdx.x * BM;
  const int nBase = blockIdx.y * BN;
  v8f z = {0.f, 0.f, 0.f, 0.f, 0.f, 0.f, 0.f, 0.f};
  v8f acc[2][2];
#pragma unroll
  for (int i = 0; i < 2; ++i)
#pragma unroll
    for (int j = 0; j < 2; ++j) acc[i][j] = z;

  stage_A_async_dense(As[0], hs, rowBase, HH, 0, wave, lane);
  load_B_f32(Bs[0], wso, 0, nBase, DD, tid);
  const int nsteps = HH / BK;
  for (int s = 0; s < nsteps; ++s) {
    const int cur = s & 1;
    cp_async_wait();
    __syncthreads();
    if (s + 1 < nsteps) {
      const int kb = (s + 1) * BK;
      stage_A_async_dense(As[cur ^ 1], hs, rowBase, HH, kb, wave, lane);
      load_B_f32(Bs[cur ^ 1], wso, kb, nBase, DD, tid);
    }
    v16bf fa0 = ld_fragA(As[cur], wm, lane);
    v16bf fa1 = ld_fragA(As[cur], wm + 16, lane);
    v16bf b0 = ld_fragB(Bs[cur], wn, lane);
    v16bf b1 = ld_fragB(Bs[cur], wn + 16, lane);
    acc[0][0] = wmma_bf16(fa0, b0, acc[0][0]);
    acc[0][1] = wmma_bf16(fa0, b1, acc[0][1]);
    acc[1][0] = wmma_bf16(fa1, b0, acc[1][0]);
    acc[1][1] = wmma_bf16(fa1, b1, acc[1][1]);
  }
  const int n_l = lane & 15, mh = (lane >> 4) * 8;
#pragma unroll
  for (int im = 0; im < 2; ++im)
#pragma unroll
    for (int in = 0; in < 2; ++in)
#pragma unroll
      for (int i = 0; i < 8; ++i) {
        size_t gm = rowBase + wm + im * 16 + mh + i;
        int gn = nBase + wn + in * 16 + n_l;
        out[gm * DD + gn] = acc[im][in][i];
      }
}

// ---------- kernel: routed stage 1 (gathered, double-buffered) ----------
__global__ __launch_bounds__(256) void k_routed_h(const unsigned short* __restrict__ xb,
                                                  const float* __restrict__ w1,
                                                  const float* __restrict__ w2,
                                                  const int* __restrict__ cnt,
                                                  const int* __restrict__ ids,
                                                  const char* __restrict__ zbuf,
                                                  unsigned short* __restrict__ hsr) {
  const int e = blockIdx.z;
  const int ne = cnt[e];
  const int tileBase = blockIdx.x * BM;
  if (tileBase >= ne) return;
  __shared__ unsigned short As[2][BM * LDT];
  __shared__ unsigned short Bs1[2][BN * LDT];
  __shared__ unsigned short Bs2[2][BN * LDT];
  __shared__ int sEnt[BM];
  const int tid = threadIdx.x, lane = tid & 31, wave = tid >> 5;
  const int wm = (wave & 3) * 32, wn = (wave >> 2) * 32;
  const int nBase = blockIdx.y * BN;
  if (tid < BM) {
    int idx = tileBase + tid;
    sEnt[tid] = (idx < ne) ? ids[e * TT + idx] : -1;
  }
  __syncthreads();
  const float* w1e = w1 + (size_t)e * DD * HH;
  const float* w2e = w2 + (size_t)e * DD * HH;
  v8f z = {0.f, 0.f, 0.f, 0.f, 0.f, 0.f, 0.f, 0.f};
  v8f a1[2][2], a2[2][2];
#pragma unroll
  for (int i = 0; i < 2; ++i)
#pragma unroll
    for (int j = 0; j < 2; ++j) { a1[i][j] = z; a2[i][j] = z; }

  stage_A_async_gather(As[0], xb, sEnt, 1, DD, 0, zbuf, wave, lane);
  load_B_f32(Bs1[0], w1e, 0, nBase, HH, tid);
  load_B_f32(Bs2[0], w2e, 0, nBase, HH, tid);
  const int nsteps = DD / BK;
  for (int s = 0; s < nsteps; ++s) {
    const int cur = s & 1;
    cp_async_wait();
    __syncthreads();
    if (s + 1 < nsteps) {
      const int kb = (s + 1) * BK;
      stage_A_async_gather(As[cur ^ 1], xb, sEnt, 1, DD, kb, zbuf, wave, lane);
      load_B_f32(Bs1[cur ^ 1], w1e, kb, nBase, HH, tid);
      load_B_f32(Bs2[cur ^ 1], w2e, kb, nBase, HH, tid);
    }
    v16bf fa0 = ld_fragA(As[cur], wm, lane);
    v16bf fa1 = ld_fragA(As[cur], wm + 16, lane);
    v16bf b10 = ld_fragB(Bs1[cur], wn, lane);
    v16bf b11 = ld_fragB(Bs1[cur], wn + 16, lane);
    v16bf b20 = ld_fragB(Bs2[cur], wn, lane);
    v16bf b21 = ld_fragB(Bs2[cur], wn + 16, lane);
    a1[0][0] = wmma_bf16(fa0, b10, a1[0][0]);
    a1[0][1] = wmma_bf16(fa0, b11, a1[0][1]);
    a1[1][0] = wmma_bf16(fa1, b10, a1[1][0]);
    a1[1][1] = wmma_bf16(fa1, b11, a1[1][1]);
    a2[0][0] = wmma_bf16(fa0, b20, a2[0][0]);
    a2[0][1] = wmma_bf16(fa0, b21, a2[0][1]);
    a2[1][0] = wmma_bf16(fa1, b20, a2[1][0]);
    a2[1][1] = wmma_bf16(fa1, b21, a2[1][1]);
  }
  const int n_l = lane & 15, mh = (lane >> 4) * 8;
#pragma unroll
  for (int im = 0; im < 2; ++im)
#pragma unroll
    for (int in = 0; in < 2; ++in)
#pragma unroll
      for (int i = 0; i < 8; ++i) {
        int rloc = wm + im * 16 + mh + i;
        int ent = sEnt[rloc];
        if (ent >= 0) {
          float h = silu_f(a1[im][in][i]) * a2[im][in][i];
          int gn = nBase + wn + in * 16 + n_l;
          hsr[(size_t)ent * HH + gn] = f2bf(h);
        }
      }
}

// ---------- kernel: routed stage 2 (gathered, double-buffered) ----------
__global__ __launch_bounds__(256) void k_routed_out(const unsigned short* __restrict__ hsr,
                                                    const float* __restrict__ wo,
                                                    const int* __restrict__ cnt,
                                                    const int* __restrict__ ids,
                                                    const char* __restrict__ zbuf,
                                                    unsigned short* __restrict__ ro) {
  const int e = blockIdx.z;
  const int ne = cnt[e];
  const int tileBase = blockIdx.x * BM;
  if (tileBase >= ne) return;
  __shared__ unsigned short As[2][BM * LDT];
  __shared__ unsigned short Bs[2][BN * LDT];
  __shared__ int sEnt[BM];
  const int tid = threadIdx.x, lane = tid & 31, wave = tid >> 5;
  const int wm = (wave & 3) * 32, wn = (wave >> 2) * 32;
  const int nBase = blockIdx.y * BN;
  if (tid < BM) {
    int idx = tileBase + tid;
    sEnt[tid] = (idx < ne) ? ids[e * TT + idx] : -1;
  }
  __syncthreads();
  const float* woe = wo + (size_t)e * HH * DD;
  v8f z = {0.f, 0.f, 0.f, 0.f, 0.f, 0.f, 0.f, 0.f};
  v8f acc[2][2];
#pragma unroll
  for (int i = 0; i < 2; ++i)
#pragma unroll
    for (int j = 0; j < 2; ++j) acc[i][j] = z;

  stage_A_async_gather(As[0], hsr, sEnt, 0, HH, 0, zbuf, wave, lane);
  load_B_f32(Bs[0], woe, 0, nBase, DD, tid);
  const int nsteps = HH / BK;
  for (int s = 0; s < nsteps; ++s) {
    const int cur = s & 1;
    cp_async_wait();
    __syncthreads();
    if (s + 1 < nsteps) {
      const int kb = (s + 1) * BK;
      stage_A_async_gather(As[cur ^ 1], hsr, sEnt, 0, HH, kb, zbuf, wave, lane);
      load_B_f32(Bs[cur ^ 1], woe, kb, nBase, DD, tid);
    }
    v16bf fa0 = ld_fragA(As[cur], wm, lane);
    v16bf fa1 = ld_fragA(As[cur], wm + 16, lane);
    v16bf b0 = ld_fragB(Bs[cur], wn, lane);
    v16bf b1 = ld_fragB(Bs[cur], wn + 16, lane);
    acc[0][0] = wmma_bf16(fa0, b0, acc[0][0]);
    acc[0][1] = wmma_bf16(fa0, b1, acc[0][1]);
    acc[1][0] = wmma_bf16(fa1, b0, acc[1][0]);
    acc[1][1] = wmma_bf16(fa1, b1, acc[1][1]);
  }
  const int n_l = lane & 15, mh = (lane >> 4) * 8;
#pragma unroll
  for (int im = 0; im < 2; ++im)
#pragma unroll
    for (int in = 0; in < 2; ++in)
#pragma unroll
      for (int i = 0; i < 8; ++i) {
        int rloc = wm + im * 16 + mh + i;
        int ent = sEnt[rloc];
        if (ent >= 0) {
          int gn = nBase + wn + in * 16 + n_l;
          ro[(size_t)ent * DD + gn] = f2bf(acc[im][in][i]);
        }
      }
}

// ---------- kernel: combine (deterministic): out[t] += gw0*ro[2t] + gw1*ro[2t+1] ----------
__global__ __launch_bounds__(256) void k_combine(float* __restrict__ out,
                                                 const unsigned short* __restrict__ ro,
                                                 const float* __restrict__ gwe) {
  size_t gid = (size_t)blockIdx.x * 256 + threadIdx.x;
  size_t base = gid * 4;
  int t = (int)(base / DD);
  int dcol = (int)(base % DD);
  float g0 = gwe[2 * t];
  float g1 = gwe[2 * t + 1];
  const uint2 r0 = *(const uint2*)(ro + (size_t)(2 * t) * DD + dcol);
  const uint2 r1 = *(const uint2*)(ro + (size_t)(2 * t + 1) * DD + dcol);
  float4 o = *(float4*)(out + base);
  o.x += g0 * bf2f((unsigned short)(r0.x & 0xffffu)) + g1 * bf2f((unsigned short)(r1.x & 0xffffu));
  o.y += g0 * bf2f((unsigned short)(r0.x >> 16))     + g1 * bf2f((unsigned short)(r1.x >> 16));
  o.z += g0 * bf2f((unsigned short)(r0.y & 0xffffu)) + g1 * bf2f((unsigned short)(r1.y & 0xffffu));
  o.w += g0 * bf2f((unsigned short)(r0.y >> 16))     + g1 * bf2f((unsigned short)(r1.y >> 16));
  *(float4*)(out + base) = o;
}

// ---------- launch ----------
extern "C" void kernel_launch(void* const* d_in, const int* in_sizes, int n_in,
                              void* d_out, int out_size, void* d_ws, size_t ws_size,
                              hipStream_t stream) {
  (void)in_sizes; (void)n_in; (void)out_size; (void)ws_size;
  const float* x   = (const float*)d_in[0];
  const float* w1  = (const float*)d_in[1];
  const float* w2  = (const float*)d_in[2];
  const float* wo  = (const float*)d_in[3];
  const float* ws1 = (const float*)d_in[4];
  const float* ws2 = (const float*)d_in[5];
  const float* wso = (const float*)d_in[6];
  const float* wr  = (const float*)d_in[7];

  float* out    = (float*)d_out;                       // [T, D]
  float* logits = out + (size_t)TT * DD;               // [T, E]

  // workspace carve-up (~151 MB)
  char* w = (char*)d_ws;
  size_t off = 0;
  int* cnt   = (int*)(w + off);             off += 256;
  char* zbuf = (char*)(w + off);            off += 256;   // 64B zeros for invalid gather rows
  int* ids   = (int*)(w + off);             off += (size_t)EE * TT * sizeof(int);
  float* gwe = (float*)(w + off);           off += (size_t)2 * TT * sizeof(float);
  unsigned short* xb  = (unsigned short*)(w + off); off += (size_t)TT * DD * 2;
  unsigned short* hs  = (unsigned short*)(w + off); off += (size_t)TT * HH * 2;
  unsigned short* hsr = (unsigned short*)(w + off); off += (size_t)2 * TT * HH * 2;
  unsigned short* ro  = (unsigned short*)(w + off); off += (size_t)2 * TT * DD * 2;

  hipMemsetAsync(w, 0, 512, stream);   // cnt + zbuf

  k_cvt_x<<<(TT * DD / 8) / 256, 256, 0, stream>>>(x, xb);
  k_router<<<TT / 256, 256, 0, stream>>>(x, wr, logits, cnt, ids, gwe);
  k_shared_h<<<dim3(TT / BM, HH / BN), 256, 0, stream>>>(xb, ws1, ws2, hs);
  k_shared_out<<<dim3(TT / BM, DD / BN), 256, 0, stream>>>(hs, wso, out);
  k_routed_h<<<dim3(TT / BM, HH / BN, EE), 256, 0, stream>>>(xb, w1, w2, cnt, ids, zbuf, hsr);
  k_routed_out<<<dim3(TT / BM, DD / BN, EE), 256, 0, stream>>>(hsr, wo, cnt, ids, zbuf, ro);
  k_combine<<<(TT * DD / 4) / 256, 256, 0, stream>>>(out, ro, gwe);
}
